// Net_23605140258866
// MI455X (gfx1250) — matrix operand order repaired and checked by reading, not compiled, not run on hardware
//
#include <hip/hip_runtime.h>
#include <hip/hip_bf16.h>

// ---------------------------------------------------------------------------
// Types for CDNA5 WMMA
// ---------------------------------------------------------------------------
typedef __attribute__((ext_vector_type(16))) __bf16 v16bf;
typedef __attribute__((ext_vector_type(8)))  float  v8f;
typedef __attribute__((ext_vector_type(4)))  int    v4i;

#define N_NODES 20000
#define N_EDGES 320000
#define MPAD    20224   // 79 * 256, zero-padded rows for guard-free staging

// ---------------------------------------------------------------------------
// Small helper kernels
// ---------------------------------------------------------------------------
__global__ void k_zero4(float4* __restrict__ p, long n4) {
    long i = (long)blockIdx.x * blockDim.x + threadIdx.x;
    if (i < n4) p[i] = make_float4(0.f, 0.f, 0.f, 0.f);
}

__global__ void k_degree(const int* __restrict__ row, const int* __restrict__ col,
                         float* __restrict__ deg, int E) {
    int e = blockIdx.x * blockDim.x + threadIdx.x;
    if (e < E) {
        int r = row[e];
        if (r != col[e]) atomicAdd(&deg[r], 1.0f);
    }
}

__global__ void k_dis(const float* __restrict__ deg, float* __restrict__ dis, int n) {
    int i = blockIdx.x * blockDim.x + threadIdx.x;
    if (i < n) {
        float d = deg[i];
        dis[i] = (d > 0.f) ? rsqrtf(d) : 0.f;
    }
}

__global__ void k_edgew(const int* __restrict__ row, const int* __restrict__ col,
                        const float* __restrict__ dis, float* __restrict__ w, int E) {
    int e = blockIdx.x * blockDim.x + threadIdx.x;
    if (e < E) {
        int r = row[e], c = col[e];
        w[e] = (r != c) ? (-dis[r] * dis[c]) : 0.f;
    }
}

// y[row[e], :] += w[e] * t[col[e], :]   (y must be pre-zeroed). One block/edge.
__global__ __launch_bounds__(256)
void k_spmm(const float* __restrict__ w, const int* __restrict__ row,
            const int* __restrict__ col, const float* __restrict__ t,
            float* __restrict__ y, int F) {
    int e = blockIdx.x;
    float we = w[e];
    if (we == 0.f) return;
    const float* ts = t + (size_t)col[e] * F;
    float*       yd = y + (size_t)row[e] * F;
    for (int f = threadIdx.x * 4; f < F; f += blockDim.x * 4) {
        float4 v = *(const float4*)(ts + f);
        atomicAdd(yd + f + 0, we * v.x);
        atomicAdd(yd + f + 1, we * v.y);
        atomicAdd(yd + f + 2, we * v.z);
        atomicAdd(yd + f + 3, we * v.w);
    }
}

// t2 = 2*t2 - t0  (Chebyshev recurrence)
__global__ void k_cheb(float4* __restrict__ t2, const float4* __restrict__ t0, long n4) {
    long i = (long)blockIdx.x * blockDim.x + threadIdx.x;
    if (i < n4) {
        float4 a = t2[i], b = t0[i];
        t2[i] = make_float4(2.f * a.x - b.x, 2.f * a.y - b.y,
                            2.f * a.z - b.z, 2.f * a.w - b.w);
    }
}

// Pack one Chebyshev basis slice (f32) into the bf16 A matrix [MPAD, Ktot].
__global__ void k_pack_a(const float* __restrict__ src, __bf16* __restrict__ A,
                         int F, int Ktot, int kofs) {
    int m = blockIdx.y;
    int f = blockIdx.x * blockDim.x + threadIdx.x;
    if (f < F)
        A[(size_t)m * Ktot + kofs + f] = (__bf16)src[(size_t)m * F + f];
}

// Pack W[K][Fin][Fout] (f32) into Bt[Fout][Ktot] (bf16), transposed.
__global__ void k_pack_w(const float* __restrict__ W, __bf16* __restrict__ Bt,
                         int Ktot, int Fout) {
    int n = blockIdx.y;
    int j = blockIdx.x * blockDim.x + threadIdx.x;  // j = k*Fin + f
    if (j < Ktot)
        Bt[(size_t)n * Ktot + j] = (__bf16)W[(size_t)j * Fout + n];
}

// ---------------------------------------------------------------------------
// Async global->LDS copy (CDNA5 GLOBAL_LOAD_ASYNC_TO_LDS_B128, ASYNCcnt)
// Builtin signature (from hipcc diagnostic): param 1 is v4i in addrspace(1).
// ---------------------------------------------------------------------------
#if defined(__gfx1250__)
__device__ __forceinline__ void async_copy_b128(const __bf16* g, __bf16* l) {
#if __has_builtin(__builtin_amdgcn_global_load_async_to_lds_b128)
    __builtin_amdgcn_global_load_async_to_lds_b128(
        (__attribute__((address_space(1))) v4i*)(void*)g,
        (__attribute__((address_space(3))) v4i*)(void*)l, 0, 0);
#else
    unsigned loff = (unsigned)(unsigned long long)
        (__attribute__((address_space(3))) void*)(void*)l;
    unsigned long long ga = (unsigned long long)g;
    asm volatile("global_load_async_to_lds_b128 %0, %1, off"
                 :: "v"(loff), "v"(ga) : "memory");
#endif
}
__device__ __forceinline__ void wait_async0() {
#if __has_builtin(__builtin_amdgcn_s_wait_asynccnt)
    __builtin_amdgcn_s_wait_asynccnt(0);
#else
    asm volatile("s_wait_asynccnt 0x0" ::: "memory");
#endif
}
#endif

// ---------------------------------------------------------------------------
// WMMA GEMM:  out = relu(A[MPAD,Ktot] @ Bt[N,Ktot]^T + bias),  f32 out, bf16 in
// Block tile 256x96, 8 waves (4 M x 2 N), wave tile 64x48, K-step 32.
// Double-buffered LDS fed by async global->LDS loads.
// ---------------------------------------------------------------------------
#define BM 256
#define BN 96
#define BK 32
#define LDSS 40  // padded LDS row stride in bf16 units (80B, 16B aligned)

union FragBF { uint4 q[2]; v16bf v; };

// A-fragment (16x32 bf16): lanes 0-15 -> K 0..7 (v0-3) & 16..23 (v4-7);
// lanes 16-31 -> K 8..15 & 24..31.  (ISA 7.12.2, 16-bit A 16x32)
__device__ __forceinline__ v16bf load_a_frag(const __bf16* s, int rowBase, int lane) {
    int r  = rowBase + (lane & 15);
    int kb = (lane >> 4) << 3;
    const __bf16* p = s + r * LDSS + kb;
    FragBF f;
    f.q[0] = *(const uint4*)(p);
    f.q[1] = *(const uint4*)(p + 16);
    return f.v;
}

// B-fragment (32x16 bf16): lanes 0-15 -> K 0..7 & 16..23; lanes 16-31 -> K 8..15 & 24..31
__device__ __forceinline__ v16bf load_b_frag(const __bf16* s, int rowBase, int lane) {
    int r  = rowBase + (lane & 15);
    int kh = (lane >> 4) << 3;
    const __bf16* p = s + r * LDSS + kh;
    FragBF f;
    f.q[0] = *(const uint4*)(p);
    f.q[1] = *(const uint4*)(p + 16);
    return f.v;
}

__global__ __launch_bounds__(256)
void k_gemm_bias_relu(const __bf16* __restrict__ A, const __bf16* __restrict__ Bt,
                      const float* __restrict__ bias, float* __restrict__ out,
                      int M, int N, int Ktot) {
    __shared__ __bf16 sA[2][BM * LDSS];
    __shared__ __bf16 sB[2][BN * LDSS];

    const int tid  = threadIdx.x;
    const int lane = tid & 31;
    const int wave = tid >> 5;
    const int wm   = wave >> 1;   // 0..3 -> M offset wm*64
    const int wn   = wave & 1;    // 0..1 -> N offset wn*48
    const int m0   = blockIdx.x * BM;
    const int n0   = blockIdx.y * BN;

    v8f acc[4][3] = {};

#if defined(__gfx1250__)
    // Stage one 256x32 A tile + 96x32 B tile into LDS buffer `buf` (async).
    auto stage = [&](int buf, int kk) {
        #pragma unroll
        for (int i = tid; i < BM * 4; i += 256) {          // 4 iters, uniform
            int r = i >> 2, c8 = (i & 3) << 3;
            async_copy_b128(A + (size_t)(m0 + r) * Ktot + kk + c8,
                            &sA[buf][r * LDSS + c8]);
        }
        for (int i = tid; i < BN * 4; i += 256) {          // 2 iters (2nd half-masked)
            int r = i >> 2, c8 = (i & 3) << 3;
            async_copy_b128(Bt + (size_t)(n0 + r) * Ktot + kk + c8,
                            &sB[buf][r * LDSS + c8]);
        }
    };

    const int nk = Ktot / BK;
    stage(0, 0);
    for (int ik = 0; ik < nk; ++ik) {
        const int buf = ik & 1;
        wait_async0();          // this wave's async fills for `buf` landed
        __syncthreads();        // everyone's did
        if (ik + 1 < nk) stage(buf ^ 1, (ik + 1) * BK);

        v16bf a0 = load_a_frag(sA[buf], wm * 64 + 0,  lane);
        v16bf a1 = load_a_frag(sA[buf], wm * 64 + 16, lane);
        v16bf a2 = load_a_frag(sA[buf], wm * 64 + 32, lane);
        v16bf a3 = load_a_frag(sA[buf], wm * 64 + 48, lane);
        v16bf b0 = load_b_frag(sB[buf], wn * 48 + 0,  lane);
        v16bf b1 = load_b_frag(sB[buf], wn * 48 + 16, lane);
        v16bf b2 = load_b_frag(sB[buf], wn * 48 + 32, lane);

        acc[0][0] = __builtin_amdgcn_wmma_f32_16x16x32_bf16(false, a0, false, b0, (short)0, acc[0][0], false, false);
        acc[0][1] = __builtin_amdgcn_wmma_f32_16x16x32_bf16(false, a0, false, b1, (short)0, acc[0][1], false, false);
        acc[0][2] = __builtin_amdgcn_wmma_f32_16x16x32_bf16(false, a0, false, b2, (short)0, acc[0][2], false, false);
        acc[1][0] = __builtin_amdgcn_wmma_f32_16x16x32_bf16(false, a1, false, b0, (short)0, acc[1][0], false, false);
        acc[1][1] = __builtin_amdgcn_wmma_f32_16x16x32_bf16(false, a1, false, b1, (short)0, acc[1][1], false, false);
        acc[1][2] = __builtin_amdgcn_wmma_f32_16x16x32_bf16(false, a1, false, b2, (short)0, acc[1][2], false, false);
        acc[2][0] = __builtin_amdgcn_wmma_f32_16x16x32_bf16(false, a2, false, b0, (short)0, acc[2][0], false, false);
        acc[2][1] = __builtin_amdgcn_wmma_f32_16x16x32_bf16(false, a2, false, b1, (short)0, acc[2][1], false, false);
        acc[2][2] = __builtin_amdgcn_wmma_f32_16x16x32_bf16(false, a2, false, b2, (short)0, acc[2][2], false, false);
        acc[3][0] = __builtin_amdgcn_wmma_f32_16x16x32_bf16(false, a3, false, b0, (short)0, acc[3][0], false, false);
        acc[3][1] = __builtin_amdgcn_wmma_f32_16x16x32_bf16(false, a3, false, b1, (short)0, acc[3][1], false, false);
        acc[3][2] = __builtin_amdgcn_wmma_f32_16x16x32_bf16(false, a3, false, b2, (short)0, acc[3][2], false, false);

        __syncthreads();        // all waves done reading `buf` before next overwrite
    }
#endif

    // Epilogue: C layout (ISA 7.12.2): VGPR v, lanes 0-15 -> M=v, N=lane;
    // lanes 16-31 -> M=v+8, N=lane-16.
    #pragma unroll
    for (int mt = 0; mt < 4; ++mt) {
        #pragma unroll
        for (int nt = 0; nt < 3; ++nt) {
            int n  = n0 + wn * 48 + nt * 16 + (lane & 15);
            float bn = bias[n];
            #pragma unroll
            for (int v = 0; v < 8; ++v) {
                int m = m0 + wm * 64 + mt * 16 + ((lane >> 4) << 3) + v;
                if (m < M) {
                    float val = acc[mt][nt][v] + bn;
                    out[(size_t)m * N + n] = val > 0.f ? val : 0.f;
                }
            }
        }
    }
}

// ---------------------------------------------------------------------------
// Host-side orchestration
// ---------------------------------------------------------------------------
static void run_layer(const float* in, int Fin, int Fout, int K,
                      const float* W, const float* bias, float* out,
                      const float* wE, const int* row, const int* col,
                      float* tb1, float* tb2, float* tb3,
                      __bf16* Abuf, __bf16* Btbuf, hipStream_t stream) {
    const int M = N_NODES, E = N_EDGES;
    const int Ktot = K * Fin;
    const long nfeat = (long)M * Fin;

    // Pack W -> Bt[Fout][Ktot] bf16
    {
        dim3 g((Ktot + 255) / 256, Fout);
        k_pack_w<<<g, 256, 0, stream>>>(W, Btbuf, Ktot, Fout);
    }
    dim3 gp((Fin + 255) / 256, M);

    // Tx0 = in
    k_pack_a<<<gp, 256, 0, stream>>>(in, Abuf, Fin, Ktot, 0);

    // Tx1 = L~ Tx0
    {
        long n4 = nfeat / 4;
        k_zero4<<<(unsigned)((n4 + 255) / 256), 256, 0, stream>>>((float4*)tb1, n4);
        k_spmm<<<E, 256, 0, stream>>>(wE, row, col, in, tb1, Fin);
        k_pack_a<<<gp, 256, 0, stream>>>(tb1, Abuf, Fin, Ktot, Fin);
    }

    // Tx_k = 2 L~ Tx_{k-1} - Tx_{k-2}
    const float* p0 = in;
    float* p1 = tb1;
    float* bufs[3] = {tb1, tb2, tb3};
    int next = 1;
    for (int k = 2; k < K; ++k) {
        float* p2 = bufs[next];
        long n4 = nfeat / 4;
        k_zero4<<<(unsigned)((n4 + 255) / 256), 256, 0, stream>>>((float4*)p2, n4);
        k_spmm<<<E, 256, 0, stream>>>(wE, row, col, p1, p2, Fin);
        k_cheb<<<(unsigned)((n4 + 255) / 256), 256, 0, stream>>>((float4*)p2, (const float4*)p0, n4);
        k_pack_a<<<gp, 256, 0, stream>>>(p2, Abuf, Fin, Ktot, k * Fin);
        p0 = p1; p1 = p2; next = (next + 1) % 3;
    }

    // Zero the padded A rows [M, MPAD) so staging needs no guards.
    {
        long padBytes = (long)(MPAD - M) * Ktot * 2;
        long n4 = padBytes / 16;
        k_zero4<<<(unsigned)((n4 + 255) / 256), 256, 0, stream>>>(
            (float4*)(Abuf + (size_t)M * Ktot), n4);
    }

    // out = relu(A @ Bt^T + bias)
    dim3 gg(MPAD / BM, Fout / BN);
    k_gemm_bias_relu<<<gg, 256, 0, stream>>>(Abuf, Btbuf, bias, out, M, Fout, Ktot);
}

extern "C" void kernel_launch(void* const* d_in, const int* in_sizes, int n_in,
                              void* d_out, int out_size, void* d_ws, size_t ws_size,
                              hipStream_t stream) {
    const int M = N_NODES, E = N_EDGES;
    const float* x   = (const float*)d_in[0];
    const int*   row = (const int*)  d_in[1];
    const int*   col = (const int*)  d_in[2];
    const float* W1  = (const float*)d_in[3];
    const float* b1  = (const float*)d_in[4];
    const float* W2  = (const float*)d_in[5];
    const float* b2  = (const float*)d_in[6];
    const float* W3  = (const float*)d_in[7];
    const float* b3  = (const float*)d_in[8];

    // Carve workspace (256B aligned slabs)
    size_t off = 0;
    char* base = (char*)d_ws;
    auto carve = [&](size_t bytes) -> void* {
        void* p = base + off;
        off += (bytes + 255) & ~(size_t)255;
        return p;
    };
    float*  deg  = (float*)carve((size_t)M * 4);
    float*  dis  = (float*)carve((size_t)M * 4);
    float*  wE   = (float*)carve((size_t)E * 4);
    float*  tb1  = (float*)carve((size_t)M * 1152 * 4);
    float*  tb2  = (float*)carve((size_t)M * 1152 * 4);
    float*  tb3  = (float*)carve((size_t)M * 1152 * 4);
    __bf16* Abuf = (__bf16*)carve((size_t)MPAD * 6912 * 2);
    __bf16* Btbf = (__bf16*)carve((size_t)1152 * 6912 * 2);
    float*  h1   = (float*)carve((size_t)M * 1152 * 4);
    float*  h2   = (float*)carve((size_t)M * 576 * 4);
    (void)ws_size; (void)in_sizes; (void)n_in; (void)out_size;

    // Graph normalization (shared by all layers)
    {
        long n4m = M / 4;  // 20000 divisible by 4
        k_zero4<<<(unsigned)((n4m + 255) / 256), 256, 0, stream>>>((float4*)deg, n4m);
        k_degree<<<(E + 255) / 256, 256, 0, stream>>>(row, col, deg, E);
        k_dis<<<(M + 255) / 256, 256, 0, stream>>>(deg, dis, M);
        k_edgew<<<(E + 255) / 256, 256, 0, stream>>>(row, col, dis, wE, E);
    }

    run_layer(x,  1152, 1152, 6, W1, b1, h1,             wE, row, col, tb1, tb2, tb3, Abuf, Btbf, stream);
    run_layer(h1, 1152,  576, 5, W2, b2, h2,             wE, row, col, tb1, tb2, tb3, Abuf, Btbf, stream);
    run_layer(h2,  576,  288, 5, W3, b3, (float*)d_out,  wE, row, col, tb1, tb2, tb3, Abuf, Btbf, stream);
}
